// VoltaGatedSelfAttention_25718264168948
// MI455X (gfx1250) — matrix operand order repaired and checked by reading, not compiled, not run on hardware
//
#include <hip/hip_runtime.h>

typedef __attribute__((ext_vector_type(16))) _Float16 v16h;
typedef __attribute__((ext_vector_type(8)))  _Float16 v8h;
typedef __attribute__((ext_vector_type(8)))  float    v8f;

#define NH    16
#define HD    64
#define HDIM  1024
#define LT    512
#define LV    256
#define LCAT  768
#define BATCH 8

static __device__ __forceinline__ v16h make_afrag(v8h c0, v8h c1) {
  v16h a;
#pragma unroll
  for (int i = 0; i < 8; ++i) { a[i] = c0[i]; a[i + 8] = c1[i]; }
  return a;
}

// ---------------- fp32 -> fp16 elementwise convert ----------------
__global__ void cvt_f16(const float* __restrict__ in, _Float16* __restrict__ out, int n) {
  int i = blockIdx.x * blockDim.x + threadIdx.x;
  if (i < n) out[i] = (_Float16)in[i];
}

// ---------------- weight transpose + convert: wt[n][k] = w[k][n] ----------------
__global__ void transpose_cvt(const float* __restrict__ w, _Float16* __restrict__ wt) {
  __shared__ float tile[32][33];
  const int tx = threadIdx.x, ty = threadIdx.y;
  const int n0 = blockIdx.x * 32, k0 = blockIdx.y * 32;
#pragma unroll
  for (int j = 0; j < 32; j += 8)
    tile[ty + j][tx] = w[(size_t)(k0 + ty + j) * HDIM + n0 + tx];
  __syncthreads();
#pragma unroll
  for (int j = 0; j < 32; j += 8)
    wt[(size_t)(n0 + ty + j) * HDIM + k0 + tx] = (_Float16)tile[tx][ty + j];
}

// ---------------- WMMA projection GEMM: out = X(f16) * W + bias ----------------
// X: [Mrows, 1024] f16 row-major.  Wt: [1024 n][1024 k] f16 (W transposed).
// vt_mode==0: out[((b*NH+h)*LCAT + soff + s)*HD + hd]   (row-major per head, for Q/K)
// vt_mode==1: out[((b*NH+h)*HD + hd)*LCAT + soff + s]   (transposed per head, for V)
__global__ __launch_bounds__(256)
void gemm_proj(const _Float16* __restrict__ X, const _Float16* __restrict__ Wt,
               const float* __restrict__ bias, _Float16* __restrict__ out,
               int lg2S, int soff, int vt_mode) {
  const int l    = threadIdx.x & 31;
  const int wid  = threadIdx.x >> 5;
  const int hlf  = l >> 4;      // 0 or 1 (lane half)
  const int ln   = l & 15;
  const int mbase = blockIdx.y * 128 + wid * 16;
  const int nbase = blockIdx.x * 64;

  v8f acc0 = {}, acc1 = {}, acc2 = {}, acc3 = {};
  const _Float16* xrow  = X + (size_t)(mbase + ln) * HDIM;
  const _Float16* wrow0 = Wt + (size_t)(nbase +  0 + ln) * HDIM;
  const _Float16* wrow1 = Wt + (size_t)(nbase + 16 + ln) * HDIM;
  const _Float16* wrow2 = Wt + (size_t)(nbase + 32 + ln) * HDIM;
  const _Float16* wrow3 = Wt + (size_t)(nbase + 48 + ln) * HDIM;
  const int ao = hlf * 16;      // A fragment K half select
  const int bo = hlf * 16;      // B fragment K half select

  // prologue: fragments for k0 = 0
  v16h a  = make_afrag(*(const v8h*)(xrow + ao), *(const v8h*)(xrow + ao + 8));
  v16h b0 = *(const v16h*)(wrow0 + bo);
  v16h b1 = *(const v16h*)(wrow1 + bo);
  v16h b2 = *(const v16h*)(wrow2 + bo);
  v16h b3 = *(const v16h*)(wrow3 + bo);

  // software-pipelined main loop: issue k0 loads before k0-32 WMMAs
#pragma unroll 1
  for (int k0 = 32; k0 < HDIM; k0 += 32) {
    v16h an  = make_afrag(*(const v8h*)(xrow + k0 + ao), *(const v8h*)(xrow + k0 + ao + 8));
    v16h bn0 = *(const v16h*)(wrow0 + k0 + bo);
    v16h bn1 = *(const v16h*)(wrow1 + k0 + bo);
    v16h bn2 = *(const v16h*)(wrow2 + k0 + bo);
    v16h bn3 = *(const v16h*)(wrow3 + k0 + bo);
    __builtin_prefetch(xrow + k0 + 32, 0, 0);
    acc0 = __builtin_amdgcn_wmma_f32_16x16x32_f16(false, a, false, b0, (short)0, acc0, false, false);
    acc1 = __builtin_amdgcn_wmma_f32_16x16x32_f16(false, a, false, b1, (short)0, acc1, false, false);
    acc2 = __builtin_amdgcn_wmma_f32_16x16x32_f16(false, a, false, b2, (short)0, acc2, false, false);
    acc3 = __builtin_amdgcn_wmma_f32_16x16x32_f16(false, a, false, b3, (short)0, acc3, false, false);
    a = an; b0 = bn0; b1 = bn1; b2 = bn2; b3 = bn3;
  }
  // epilogue step (k0 = HDIM-32 fragments)
  acc0 = __builtin_amdgcn_wmma_f32_16x16x32_f16(false, a, false, b0, (short)0, acc0, false, false);
  acc1 = __builtin_amdgcn_wmma_f32_16x16x32_f16(false, a, false, b1, (short)0, acc1, false, false);
  acc2 = __builtin_amdgcn_wmma_f32_16x16x32_f16(false, a, false, b2, (short)0, acc2, false, false);
  acc3 = __builtin_amdgcn_wmma_f32_16x16x32_f16(false, a, false, b3, (short)0, acc3, false, false);

  const int Smask = (1 << lg2S) - 1;
  v8f accs[4] = {acc0, acc1, acc2, acc3};
#pragma unroll
  for (int t = 0; t < 4; ++t) {
    const int c  = nbase + t * 16 + ln;     // output feature column
    const float bv = bias[c];
    const int h = c >> 6, hd = c & 63;
#pragma unroll
    for (int r = 0; r < 8; ++r) {
      const int token = mbase + r + hlf * 8;           // D layout: lanes16-31 = M+8
      const int bb = token >> lg2S;
      const int s  = token & Smask;
      const float v = accs[t][r] + bv;
      size_t addr;
      if (vt_mode)
        addr = ((size_t)(bb * NH + h) * HD + hd) * LCAT + (size_t)(soff + s);
      else
        addr = ((size_t)(bb * NH + h) * LCAT + (size_t)(soff + s)) * HD + hd;
      out[addr] = (_Float16)v;
    }
  }
}

// ---------------- fused flash attention over concatenated K/V (768 keys) ----------------
// Q:  [bh][768][64] f16   Kc: [bh][768][64] f16   Vt: [bh][64][768] f16
// out: t_ctx [8,512,1024] f32 then v_ctx [8,256,1024] f32 (flat concat)
__global__ __launch_bounds__(128)
void attn(const _Float16* __restrict__ Q, const _Float16* __restrict__ Kc,
          const _Float16* __restrict__ Vt, const float* __restrict__ tmask,
          const float* __restrict__ vmask, float* __restrict__ out) {
  __shared__ _Float16 Pls[4][16 * 32];     // per-wave P staging (D-layout -> A-layout)
  const int l   = threadIdx.x & 31;
  const int wid = threadIdx.x >> 5;
  const int hlf = l >> 4;
  const int ln  = l & 15;
  const int bh  = blockIdx.y;
  const int b   = bh >> 4, h = bh & 15;
  const int qbase = (blockIdx.x * 4 + wid) * 16;
  _Float16* P = Pls[wid];

  // Q A-fragments for hd-chunks [0,32) and [32,64)
  const _Float16* qrow = Q + ((size_t)bh * LCAT + (size_t)(qbase + ln)) * HD;
  const int ao = hlf * 16;
  v16h a0 = make_afrag(*(const v8h*)(qrow + ao),      *(const v8h*)(qrow + ao + 8));
  v16h a1 = make_afrag(*(const v8h*)(qrow + 32 + ao), *(const v8h*)(qrow + 32 + ao + 8));

  v8f o0 = {}, o1 = {}, o2 = {}, o3 = {};
  float mrow[8], lrow[8];
#pragma unroll
  for (int r = 0; r < 8; ++r) { mrow[r] = -3.0e38f; lrow[r] = 0.0f; }

  const _Float16* kbase = Kc + (size_t)bh * LCAT * HD;
  const _Float16* vbase = Vt + (size_t)bh * HD * LCAT;

#pragma unroll 1
  for (int kb = 0; kb < LCAT; kb += 32) {
    // ---- scores: S = Q * K^T for 32 keys (two 16-key N tiles, HD split in two K steps)
    v8f s0 = {}, s1 = {};
    {
      const _Float16* kr0 = kbase + (size_t)(kb + ln) * HD + hlf * 16;
      const _Float16* kr1 = kbase + (size_t)(kb + 16 + ln) * HD + hlf * 16;
      v16h bk00 = *(const v16h*)(kr0);
      v16h bk01 = *(const v16h*)(kr0 + 32);
      v16h bk10 = *(const v16h*)(kr1);
      v16h bk11 = *(const v16h*)(kr1 + 32);
      __builtin_prefetch(kr0 + 32 * HD, 0, 0);   // next key tile
      s0 = __builtin_amdgcn_wmma_f32_16x16x32_f16(false, a0, false, bk00, (short)0, s0, false, false);
      s0 = __builtin_amdgcn_wmma_f32_16x16x32_f16(false, a1, false, bk01, (short)0, s0, false, false);
      s1 = __builtin_amdgcn_wmma_f32_16x16x32_f16(false, a0, false, bk10, (short)0, s1, false, false);
      s1 = __builtin_amdgcn_wmma_f32_16x16x32_f16(false, a1, false, bk11, (short)0, s1, false, false);
    }
    // ---- issue V B-fragment loads early: softmax VALU work hides their latency
    const int vo = kb + hlf * 16;
    v16h bv0 = *(const v16h*)(vbase + (size_t)(0  + ln) * LCAT + vo);
    v16h bv1 = *(const v16h*)(vbase + (size_t)(16 + ln) * LCAT + vo);
    v16h bv2 = *(const v16h*)(vbase + (size_t)(32 + ln) * LCAT + vo);
    v16h bv3 = *(const v16h*)(vbase + (size_t)(48 + ln) * LCAT + vo);
    // ---- scale + mask (per-lane key column in D layout)
    const int kn0 = kb + ln, kn1 = kn0 + 16;
    const float msk0 = (kn0 < LT) ? tmask[b * LT + kn0] : vmask[b * LV + (kn0 - LT)];
    const float msk1 = (kn1 < LT) ? tmask[b * LT + kn1] : vmask[b * LV + (kn1 - LT)];
    // ---- online softmax: rows live across 16-lane groups; shfl_xor stays in-group
    float fs[8];
#pragma unroll
    for (int r = 0; r < 8; ++r) {
      float x0 = s0[r] * 0.125f + msk0;      // 1/sqrt(64)
      float x1 = s1[r] * 0.125f + msk1;
      float rmax = fmaxf(x0, x1);
#pragma unroll
      for (int off = 8; off > 0; off >>= 1) rmax = fmaxf(rmax, __shfl_xor(rmax, off, 32));
      float nm = fmaxf(mrow[r], rmax);
      fs[r] = __expf(mrow[r] - nm);
      mrow[r] = nm;
      float p0 = __expf(x0 - nm);
      float p1 = __expf(x1 - nm);
      float ts = p0 + p1;
#pragma unroll
      for (int off = 8; off > 0; off >>= 1) ts += __shfl_xor(ts, off, 32);
      lrow[r] = lrow[r] * fs[r] + ts;
      s0[r] = p0; s1[r] = p1;
    }
#pragma unroll
    for (int r = 0; r < 8; ++r) { o0[r] *= fs[r]; o1[r] *= fs[r]; o2[r] *= fs[r]; o3[r] *= fs[r]; }
    // ---- stage P through LDS: D-fragment -> row-major 16x32
#pragma unroll
    for (int r = 0; r < 8; ++r) {
      const int m = r + hlf * 8;
      P[m * 32 + ln]      = (_Float16)s0[r];
      P[m * 32 + 16 + ln] = (_Float16)s1[r];
    }
    // ---- reload P as an A-fragment (M=query, K=32 keys)
    const _Float16* prow = P + ln * 32 + hlf * 16;
    v16h pa = make_afrag(*(const v8h*)(prow), *(const v8h*)(prow + 8));
    // ---- O += P * V  (V^T rows contiguous -> straight B-fragment loads)
    o0 = __builtin_amdgcn_wmma_f32_16x16x32_f16(false, pa, false, bv0, (short)0, o0, false, false);
    o1 = __builtin_amdgcn_wmma_f32_16x16x32_f16(false, pa, false, bv1, (short)0, o1, false, false);
    o2 = __builtin_amdgcn_wmma_f32_16x16x32_f16(false, pa, false, bv2, (short)0, o2, false, false);
    o3 = __builtin_amdgcn_wmma_f32_16x16x32_f16(false, pa, false, bv3, (short)0, o3, false, false);
  }

  // ---- normalize and scatter to merged-head fp32 output
#pragma unroll
  for (int r = 0; r < 8; ++r) {
    const float inv = 1.0f / lrow[r];
    const int m = r + hlf * 8;
    const int q = qbase + m;
    size_t rowaddr;
    if (q < LT)
      rowaddr = (size_t)(b * LT + q) * HDIM + h * HD;
    else
      rowaddr = (size_t)BATCH * LT * HDIM + (size_t)(b * LV + (q - LT)) * HDIM + h * HD;
    float* op = out + rowaddr + ln;
    op[0]  = o0[r] * inv;
    op[16] = o1[r] * inv;
    op[32] = o2[r] * inv;
    op[48] = o3[r] * inv;
  }
}

extern "C" void kernel_launch(void* const* d_in, const int* in_sizes, int n_in,
                              void* d_out, int out_size, void* d_ws, size_t ws_size,
                              hipStream_t stream) {
  (void)in_sizes; (void)n_in; (void)out_size; (void)ws_size;
  const float* t_hidden = (const float*)d_in[0];
  const float* v_hidden = (const float*)d_in[1];
  const float* t_mask   = (const float*)d_in[2];
  const float* v_mask   = (const float*)d_in[3];
  const float* w[6]    = {(const float*)d_in[4], (const float*)d_in[5], (const float*)d_in[6],
                          (const float*)d_in[7], (const float*)d_in[8], (const float*)d_in[9]};
  const float* bias[6] = {(const float*)d_in[10], (const float*)d_in[11], (const float*)d_in[12],
                          (const float*)d_in[13], (const float*)d_in[14], (const float*)d_in[15]};

  // workspace layout (f16 elems)
  _Float16* ws = (_Float16*)d_ws;
  _Float16* Xt = ws;                          // 8*512*1024   = 4194304
  _Float16* Xv = Xt + 4194304;                // 8*256*1024   = 2097152
  _Float16* Wt = Xv + 2097152;                // 6 * 1048576
  _Float16* Qc = Wt + 6 * 1048576;            // 8*16*768*64  = 6291456
  _Float16* Kc = Qc + 6291456;
  _Float16* Vt = Kc + 6291456;

  cvt_f16<<<4194304 / 256, 256, 0, stream>>>(t_hidden, Xt, 4194304);
  cvt_f16<<<2097152 / 256, 256, 0, stream>>>(v_hidden, Xv, 2097152);

  dim3 tb(32, 8), tg(32, 32);
  for (int i = 0; i < 6; ++i)
    transpose_cvt<<<tg, tb, 0, stream>>>(w[i], Wt + (size_t)i * 1048576);

  dim3 gb(256);
  dim3 gt(16, 32);   // N=1024/64 tiles x, M=4096/128 tiles y
  dim3 gv(16, 16);   // M=2048/128
  // text projections (lg2S=9, soff=0)
  gemm_proj<<<gt, gb, 0, stream>>>(Xt, Wt + 0 * 1048576, bias[0], Qc, 9, 0, 0);
  gemm_proj<<<gt, gb, 0, stream>>>(Xt, Wt + 1 * 1048576, bias[1], Kc, 9, 0, 0);
  gemm_proj<<<gt, gb, 0, stream>>>(Xt, Wt + 2 * 1048576, bias[2], Vt, 9, 0, 1);
  // video projections (lg2S=8, soff=512)
  gemm_proj<<<gv, gb, 0, stream>>>(Xv, Wt + 3 * 1048576, bias[3], Qc, 8, 512, 0);
  gemm_proj<<<gv, gb, 0, stream>>>(Xv, Wt + 4 * 1048576, bias[4], Kc, 8, 512, 0);
  gemm_proj<<<gv, gb, 0, stream>>>(Xv, Wt + 5 * 1048576, bias[5], Vt, 8, 512, 1);

  // fused flash attention: 48 query tiles x 128 (b,h), 4 waves/block
  attn<<<dim3(12, 128), dim3(128), 0, stream>>>(Qc, Kc, Vt, t_mask, v_mask, (float*)d_out);
}
